// Mamba_FFT_75703093559717
// MI455X (gfx1250) — matrix operand order repaired and checked by reading, not compiled, not run on hardware
//
#include <hip/hip_runtime.h>
#include <stdint.h>
#include <math.h>

// Problem constants (from reference setup_inputs)
#define BN   2
#define LSEQ 1024
#define DMOD 1024
#define DIN  2048
#define DST  16
#define DRK  64

typedef __attribute__((ext_vector_type(16))) __bf16       bf16x16;
typedef __attribute__((ext_vector_type(8)))  float        f32x8;
typedef __attribute__((ext_vector_type(8)))  unsigned int u32x8;

__device__ __forceinline__ unsigned short bfbits(float f) {
  union { float f; unsigned u; } v; v.f = f;
  unsigned u = v.u;
  u += 0x7FFFu + ((u >> 16) & 1u);            // round-to-nearest-even
  return (unsigned short)(u >> 16);
}
__device__ __forceinline__ unsigned pack2bf(float a, float b) {
  return (unsigned)bfbits(a) | ((unsigned)bfbits(b) << 16);
}
__device__ __forceinline__ float silu_f(float x) {
  return x / (1.0f + __expf(-x));
}

// ---------------------------------------------------------------------------
// WMMA GEMM: C[m,n] (+)= sum_k A[m,k]*B[k,n]; fp32 in/out, bf16 MACs.
// 128x64 tile, 256 threads (8 waves), double-buffered LDS, K step 32.
// Per-thread staging coordinates are hoisted; only k0 varies per call.
// ---------------------------------------------------------------------------

// Stage A tile (128 x 32). thread t owns (arow, akb..akb+15); ag = clamped row.
__device__ __forceinline__ void stage_A(__bf16 (*__restrict__ As)[40],
                                        const float* __restrict__ ag,
                                        int arow, int akb,
                                        int k0, int K, int lak)
{
  float v[16];
  int krem = K - k0;
  if (lak == 1) {
    const float* a0 = ag + k0 + akb;
    if (krem >= 32) {                        // fast: 4x float4
      float4 f0 = ((const float4*)a0)[0];
      float4 f1 = ((const float4*)a0)[1];
      float4 f2 = ((const float4*)a0)[2];
      float4 f3 = ((const float4*)a0)[3];
      v[0]=f0.x; v[1]=f0.y; v[2]=f0.z; v[3]=f0.w;
      v[4]=f1.x; v[5]=f1.y; v[6]=f1.z; v[7]=f1.w;
      v[8]=f2.x; v[9]=f2.y; v[10]=f2.z; v[11]=f2.w;
      v[12]=f3.x; v[13]=f3.y; v[14]=f3.z; v[15]=f3.w;
    } else {
#pragma unroll
      for (int j = 0; j < 16; ++j) {
        int kk = akb + j;
        float x = ag[k0 + (kk < krem ? kk : krem - 1)];
        v[j] = kk < krem ? x : 0.0f;
      }
    }
  } else if (krem >= 32) {                   // strided fast: pointer walk
    const float* a0 = ag + (size_t)(k0 + akb) * lak;
#pragma unroll
    for (int j = 0; j < 16; ++j) { v[j] = *a0; a0 += lak; }
  } else {                                   // strided edge (clamped)
#pragma unroll
    for (int j = 0; j < 16; ++j) {
      int kk = akb + j;
      int ck = kk < krem ? kk : krem - 1;
      float x = ag[(size_t)(k0 + ck) * lak];
      v[j] = kk < krem ? x : 0.0f;
    }
  }
  uint4 q0, q1;
  q0.x = pack2bf(v[0], v[1]);  q0.y = pack2bf(v[2], v[3]);
  q0.z = pack2bf(v[4], v[5]);  q0.w = pack2bf(v[6], v[7]);
  q1.x = pack2bf(v[8], v[9]);  q1.y = pack2bf(v[10], v[11]);
  q1.z = pack2bf(v[12], v[13]); q1.w = pack2bf(v[14], v[15]);
  *(uint4*)&As[arow][akb]     = q0;
  *(uint4*)&As[arow][akb + 8] = q1;
}

// Stage B tile (32 x 64) into Bs[n][k]. thread t owns (bn, bkb..bkb+7).
__device__ __forceinline__ void stage_B(__bf16 (*__restrict__ Bs)[40],
                                        const float* __restrict__ bg,
                                        int bn, int bkb,
                                        int k0, int K, int lbk)
{
  float v[8];
  int krem = K - k0;
  if (lbk == 1) {
    const float* b0 = bg + k0 + bkb;
    if (krem >= 32) {                        // fast: 2x float4
      float4 f0 = ((const float4*)b0)[0];
      float4 f1 = ((const float4*)b0)[1];
      v[0]=f0.x; v[1]=f0.y; v[2]=f0.z; v[3]=f0.w;
      v[4]=f1.x; v[5]=f1.y; v[6]=f1.z; v[7]=f1.w;
    } else {
#pragma unroll
      for (int j = 0; j < 8; ++j) {
        int kk = bkb + j;
        float x = bg[k0 + (kk < krem ? kk : krem - 1)];
        v[j] = kk < krem ? x : 0.0f;
      }
    }
  } else if (krem >= 32) {                   // strided fast: pointer walk,
    const float* b0 = bg + (size_t)(k0 + bkb) * lbk;   // coalesced across n
#pragma unroll
    for (int r = 0; r < 8; ++r) { v[r] = *b0; b0 += lbk; }
  } else {                                   // strided edge (clamped)
#pragma unroll
    for (int r = 0; r < 8; ++r) {
      int kk = bkb + r;
      int ck = kk < krem ? kk : krem - 1;
      float x = bg[(size_t)(k0 + ck) * lbk];
      v[r] = kk < krem ? x : 0.0f;
    }
  }
  uint4 q;
  q.x = pack2bf(v[0], v[1]); q.y = pack2bf(v[2], v[3]);
  q.z = pack2bf(v[4], v[5]); q.w = pack2bf(v[6], v[7]);
  *(uint4*)&Bs[bn][bkb] = q;
}

__global__ __launch_bounds__(256)
void wmma_gemm(const float* __restrict__ A, const float* __restrict__ B,
               float* __restrict__ C, int M, int N, int K,
               long long aB, long long bB, long long cB,
               int lam, int lak, int lbk, int lbn, int lcm, int lcn,
               int accumulate)
{
  __shared__ __align__(16) __bf16 As[2][128][40];  // 20 KB
  __shared__ __align__(16) __bf16 Bs[2][64][40];   // 10 KB

  const int m0   = blockIdx.y * 128;
  const int n0   = blockIdx.x * 64;
  const int t    = threadIdx.x;
  const int lane = t & 31;
  const int wv   = t >> 5;

  const float* Ab = A + (size_t)blockIdx.z * aB;
  const float* Bb = B + (size_t)blockIdx.z * bB;
  float*       Cb = C + (size_t)blockIdx.z * cB;

  // hoisted per-thread staging coordinates (row/col clamped once)
  const int arow = t >> 1;
  const int akb  = (t & 1) << 4;
  int cm = m0 + arow; cm = cm < M ? cm : (M - 1);
  const float* aRow = Ab + (size_t)cm * lam;

  const int bn  = (lbk == 1) ? (t >> 2) : (t & 63);
  const int bkb = (lbk == 1) ? ((t & 3) << 3) : ((t >> 6) << 3);
  int cn = n0 + bn; cn = cn < N ? cn : (N - 1);
  const float* bCol = Bb + (size_t)cn * lbn;

  f32x8 acc[4];
#pragma unroll
  for (int j = 0; j < 4; ++j)
#pragma unroll
    for (int v = 0; v < 8; ++v) acc[j][v] = 0.0f;

  stage_A(As[0], aRow, arow, akb, 0, K, lak);
  stage_B(Bs[0], bCol, bn, bkb, 0, K, lbk);

  int p = 0;
  for (int k0 = 0; k0 < K; k0 += 32) {
    __syncthreads();
    if (k0 + 32 < K) {                     // prefetch next tile into other buf
      stage_A(As[p ^ 1], aRow, arow, akb, k0 + 32, K, lak);
      stage_B(Bs[p ^ 1], bCol, bn, bkb, k0 + 32, K, lbk);
    }

    // A fragment: row = wv*16 + lane%16; two 8-elt K runs at base, base+16
    const __bf16* ap = &As[p][wv * 16 + (lane & 15)][(lane >> 4) << 3];
    uint4 qa0 = *(const uint4*)ap;
    uint4 qa1 = *(const uint4*)(ap + 16);
    u32x8 araw;
    araw[0] = qa0.x; araw[1] = qa0.y; araw[2] = qa0.z; araw[3] = qa0.w;
    araw[4] = qa1.x; araw[5] = qa1.y; araw[6] = qa1.z; araw[7] = qa1.w;
    bf16x16 afrag = __builtin_bit_cast(bf16x16, araw);

#pragma unroll
    for (int j = 0; j < 4; ++j) {
      const __bf16* bp = &Bs[p][j * 16 + (lane & 15)][(lane >> 4) << 3];
      uint4 qb0 = *(const uint4*)bp;
      uint4 qb1 = *(const uint4*)(bp + 16);
      u32x8 braw;
      braw[0] = qb0.x; braw[1] = qb0.y; braw[2] = qb0.z; braw[3] = qb0.w;
      braw[4] = qb1.x; braw[5] = qb1.y; braw[6] = qb1.z; braw[7] = qb1.w;
      bf16x16 bfrag = __builtin_bit_cast(bf16x16, braw);

      acc[j] = __builtin_amdgcn_wmma_f32_16x16x32_bf16(
          false, afrag, false, bfrag, (short)0, acc[j], false, false);
    }
    p ^= 1;
  }

  // Epilogue: lanes 0-15 -> M=v, lanes 16-31 -> M=v+8 ; N = lane%16
#pragma unroll
  for (int j = 0; j < 4; ++j) {
#pragma unroll
    for (int v = 0; v < 8; ++v) {
      int mm = m0 + wv * 16 + v + ((lane >> 4) << 3);
      int nn = n0 + j * 16 + (lane & 15);
      if (mm < M && nn < N) {
        size_t ci = (size_t)mm * lcm + (size_t)nn * lcn;
        float r = acc[j][v];
        if (accumulate) r += Cb[ci];
        Cb[ci] = r;
      }
    }
  }
}

// ---------------------------------------------------------------------------
// Elementwise / scan / FFT kernels
// ---------------------------------------------------------------------------
__global__ void pack_w(const float* __restrict__ w, float* __restrict__ wp, int total)
{
  int idx = blockIdx.x * 256 + threadIdx.x;
  if (idx >= total) return;                 // layout wp[tap][o][i]
  int i   = idx & (2 * DIN - 1);
  int o   = (idx / (2 * DIN)) & (2 * DIN - 1);
  int tap = idx / (2 * DIN * 2 * DIN);
  wp[idx] = w[((size_t)o * (2 * DIN) + i) * 5 + tap];
}

__global__ void reverse_l(const float* __restrict__ src, float* __restrict__ dst, int total)
{
  int idx = blockIdx.x * 256 + threadIdx.x;
  if (idx >= total) return;
  int l = idx % LSEQ;
  int r = idx / LSEQ;
  dst[(size_t)r * LSEQ + l] = src[(size_t)r * LSEQ + (LSEQ - 1 - l)];
}

__global__ void dwconv_silu(const float* __restrict__ xz, const float* __restrict__ w,
                            const float* __restrict__ bias, float* __restrict__ out, int total)
{
  int idx = blockIdx.x * 256 + threadIdx.x;
  if (idx >= total) return;
  int l = idx % LSEQ;
  int d = (idx / LSEQ) % DIN;
  int b = idx / (LSEQ * DIN);
  const float* xp = xz + (size_t)b * (2LL * DIN * LSEQ) + (size_t)d * LSEQ;
  float s = bias[d];
#pragma unroll
  for (int k = 0; k < 4; ++k) {
    int li = l - 3 + k;
    if (li >= 0) s += xp[li] * w[d * 4 + k];
  }
  out[idx] = silu_f(s);
}

__global__ __launch_bounds__(128)
void selective_scan(const float* __restrict__ dpre, const float* __restrict__ dtb,
                    const float* __restrict__ Alog, const float* __restrict__ xdbl,
                    const float* __restrict__ xc, const float* __restrict__ Dp,
                    const float* __restrict__ xz, float* __restrict__ y)
{
  int t = blockIdx.x * 128 + threadIdx.x;
  if (t >= BN * DIN) return;
  int b = t / DIN, d = t % DIN;
  float Ar[DST], h[DST];
#pragma unroll
  for (int n = 0; n < DST; ++n) { Ar[n] = -__expf(Alog[d * DST + n]); h[n] = 0.0f; }
  const float db = dtb[d];
  const float Dv = Dp[d];
  const float* dp = dpre + ((size_t)b * DIN + d) * LSEQ;
  const float* xp = xc   + ((size_t)b * DIN + d) * LSEQ;
  const float* zp = xz + (size_t)b * (2LL * DIN * LSEQ) + (size_t)(DIN + d) * LSEQ;
  const float* Bp = xdbl + (size_t)b * ((DRK + 2 * DST) * LSEQ) + (size_t)DRK * LSEQ;
  const float* Cp = Bp + (size_t)DST * LSEQ;
  float* yp = y + ((size_t)b * DIN + d) * LSEQ;

  for (int l = 0; l < LSEQ; ++l) {
    float pre   = dp[l] + db;
    float delta = (pre > 20.0f) ? pre : log1pf(__expf(pre));
    float xv    = xp[l];
    float dx    = delta * xv;
    float accv  = 0.0f;
#pragma unroll
    for (int n = 0; n < DST; ++n) {
      h[n]  = __expf(delta * Ar[n]) * h[n] + dx * Bp[(size_t)n * LSEQ + l];
      accv += h[n] * Cp[(size_t)n * LSEQ + l];
    }
    accv += Dv * xv;
    float zv = zp[l];
    yp[l] = accv * silu_f(zv);
  }
}

template <int N, int LOG2N>
__device__ void fft_shared(float* sre, float* sim, int tid, int nthr, bool inverse)
{
  for (int i = tid; i < N; i += nthr) {
    unsigned j = __brev((unsigned)i) >> (32 - LOG2N);
    if (j > (unsigned)i) {
      float tr = sre[i]; sre[i] = sre[j]; sre[j] = tr;
      float ti = sim[i]; sim[i] = sim[j]; sim[j] = ti;
    }
  }
  __syncthreads();
  for (int len = 2; len <= N; len <<= 1) {
    int half = len >> 1;
    float ang = (inverse ? 6.283185307179586f : -6.283185307179586f) / (float)len;
    for (int bt = tid; bt < (N >> 1); bt += nthr) {
      int grp = bt / half;
      int pos = bt - grp * half;
      int i0 = grp * len + pos;
      int i1 = i0 + half;
      float s, c;
      __sincosf(ang * (float)pos, &s, &c);
      float xr = sre[i1], xi = sim[i1];
      float tr = c * xr - s * xi;
      float ti = c * xi + s * xr;
      float ur = sre[i0], ui = sim[i0];
      sre[i1] = ur - tr; sim[i1] = ui - ti;
      sre[i0] = ur + tr; sim[i0] = ui + ti;
    }
    __syncthreads();
  }
}

__global__ __launch_bounds__(256)
void fft_rows_r2c(const float* __restrict__ xz, float* __restrict__ fre, float* __restrict__ fim)
{
  __shared__ float sre[LSEQ];
  __shared__ float sim[LSEQ];
  int row = blockIdx.x;                  // b*DIN + d
  int b = row / DIN, d = row % DIN;
  const float* src = xz + (size_t)b * (2LL * DIN * LSEQ) + (size_t)d * LSEQ;
  for (int i = threadIdx.x; i < LSEQ; i += 256) { sre[i] = src[i]; sim[i] = 0.0f; }
  __syncthreads();
  fft_shared<LSEQ, 10>(sre, sim, threadIdx.x, 256, false);
  size_t base = (size_t)row * LSEQ;
  for (int i = threadIdx.x; i < LSEQ; i += 256) { fre[base + i] = sre[i]; fim[base + i] = sim[i]; }
}

__global__ __launch_bounds__(256)
void fft_rows_c2c(float* __restrict__ fre, float* __restrict__ fim, int inverse)
{
  __shared__ float sre[LSEQ];
  __shared__ float sim[LSEQ];
  int row = blockIdx.x;
  size_t base = (size_t)row * LSEQ;
  for (int i = threadIdx.x; i < LSEQ; i += 256) { sre[i] = fre[base + i]; sim[i] = fim[base + i]; }
  __syncthreads();
  fft_shared<LSEQ, 10>(sre, sim, threadIdx.x, 256, inverse != 0);
  float sc = inverse ? (1.0f / LSEQ) : 1.0f;
  for (int i = threadIdx.x; i < LSEQ; i += 256) { fre[base + i] = sre[i] * sc; fim[base + i] = sim[i] * sc; }
}

__global__ __launch_bounds__(256)
void fft_cols(float* __restrict__ fre, float* __restrict__ fim, int inverse)
{
  __shared__ float sre[DIN];             // 2048 pts = 16KB LDS (of 320KB/WGP)
  __shared__ float sim[DIN];
  int col = blockIdx.x;                  // b*LSEQ + l
  int b = col / LSEQ, l = col % LSEQ;
  size_t base = (size_t)b * DIN * LSEQ + l;
  for (int i = threadIdx.x; i < DIN; i += 256) {
    sre[i] = fre[base + (size_t)i * LSEQ];
    sim[i] = fim[base + (size_t)i * LSEQ];
  }
  __syncthreads();
  fft_shared<DIN, 11>(sre, sim, threadIdx.x, 256, inverse != 0);
  float sc = inverse ? (1.0f / DIN) : 1.0f;
  for (int i = threadIdx.x; i < DIN; i += 256) {
    fre[base + (size_t)i * LSEQ] = sre[i] * sc;
    fim[base + (size_t)i * LSEQ] = sim[i] * sc;
  }
}

__global__ void build_xcpad(const float* __restrict__ fre, const float* __restrict__ fim,
                            float* __restrict__ xcpad, int total)
{
  int idx = blockIdx.x * 256 + threadIdx.x;
  if (idx >= total) return;
  const int LP = LSEQ + 4;
  int lp = idx % LP;
  int c  = (idx / LP) % (2 * DIN);
  int b  = idx / (LP * 2 * DIN);
  float v = 0.0f;
  int l = lp - 2;
  if (l >= 0 && l < LSEQ) {
    if (c < DIN) v = fre[((size_t)b * DIN + c) * LSEQ + l];
    else         v = fim[((size_t)b * DIN + (c - DIN)) * LSEQ + l];
  }
  xcpad[idx] = v;
}

__global__ void bias_silu_split(const float* __restrict__ conv, const float* __restrict__ bias,
                                float* __restrict__ fre, float* __restrict__ fim, int total)
{
  int idx = blockIdx.x * 256 + threadIdx.x;
  if (idx >= total) return;               // total = BN*DIN*LSEQ
  int l = idx % LSEQ;
  int c = (idx / LSEQ) % DIN;
  int b = idx / (LSEQ * DIN);
  size_t cb = (size_t)b * (2LL * DIN * LSEQ);
  float re = conv[cb + (size_t)c * LSEQ + l] + bias[c];
  float im = conv[cb + (size_t)(c + DIN) * LSEQ + l] + bias[c + DIN];
  fre[idx] = silu_f(re);
  fim[idx] = silu_f(im);
}

__global__ void zmax_kernel(const float* __restrict__ xz, float* __restrict__ zmax)
{
  int t = blockIdx.x * 128 + threadIdx.x;
  if (t >= BN * DIN) return;
  int b = t / DIN, d = t % DIN;
  const float* zp = xz + (size_t)b * (2LL * DIN * LSEQ) + (size_t)(DIN + d) * LSEQ;
  float m = -INFINITY;
  for (int l = 0; l < LSEQ; ++l) m = fmaxf(m, silu_f(zp[l]));
  zmax[t] = m;
}

__global__ __launch_bounds__(256)
void combine_norm(const float* __restrict__ yf, const float* __restrict__ yb,
                  const float* __restrict__ xre, const float* __restrict__ xim,
                  const float* __restrict__ zmax, float* __restrict__ ytot)
{
  __shared__ float s1[256];
  __shared__ float s2[256];
  int row = blockIdx.x;                  // b*DIN + d
  int tid = threadIdx.x;
  const float zm = fabsf(zmax[row]);
  size_t base = (size_t)row * LSEQ;
  float vloc[4];
  float lsum = 0.0f, lsq = 0.0f;
#pragma unroll
  for (int i = 0; i < 4; ++i) {
    int l = tid + i * 256;
    float re = xre[base + l], im = xim[base + l];
    float v  = sqrtf(re * re + im * im) * zm;
    vloc[i] = v;
    lsum += v; lsq += v * v;
  }
  s1[tid] = lsum; s2[tid] = lsq;
  __syncthreads();
  for (int s = 128; s > 0; s >>= 1) {
    if (tid < s) { s1[tid] += s1[tid + s]; s2[tid] += s2[tid + s]; }
    __syncthreads();
  }
  float mu  = s1[0] * (1.0f / LSEQ);
  float var = s2[0] * (1.0f / LSEQ) - mu * mu;
  float inv = rsqrtf(var + 1e-5f);
#pragma unroll
  for (int i = 0; i < 4; ++i) {
    int l = tid + i * 256;
    ytot[base + l] = yf[base + l] + yb[base + (LSEQ - 1 - l)] + (vloc[i] - mu) * inv;
  }
}

// ---------------------------------------------------------------------------
// Orchestration
// ---------------------------------------------------------------------------
static void launch_gemm(hipStream_t s, const float* A, const float* B, float* C,
                        int M, int N, int K, long long aB, long long bB, long long cB,
                        int lam, int lak, int lbk, int lbn, int lcm, int lcn,
                        int accumulate)
{
  dim3 g((N + 63) / 64, (M + 127) / 128, BN);
  wmma_gemm<<<g, 256, 0, s>>>(A, B, C, M, N, K, aB, bB, cB,
                              lam, lak, lbk, lbn, lcm, lcn, accumulate);
}

extern "C" void kernel_launch(void* const* d_in, const int* in_sizes, int n_in,
                              void* d_out, int out_size, void* d_ws, size_t ws_size,
                              hipStream_t stream)
{
  const float* hidden   = (const float*)d_in[0];
  const float* inproj_w = (const float*)d_in[1];
  const float* conv_w   = (const float*)d_in[2];
  const float* conv_b   = (const float*)d_in[3];
  const float* xproj_w  = (const float*)d_in[4];
  const float* dtproj_w = (const float*)d_in[5];
  const float* dtproj_b = (const float*)d_in[6];
  const float* A_log    = (const float*)d_in[7];
  const float* Dp       = (const float*)d_in[8];
  const float* convb_w  = (const float*)d_in[9];
  const float* convb_b  = (const float*)d_in[10];
  const float* xprojb_w = (const float*)d_in[11];
  const float* dtprojb_w= (const float*)d_in[12];
  const float* dtprojb_b= (const float*)d_in[13];
  const float* Ab_log   = (const float*)d_in[14];
  const float* Dbp      = (const float*)d_in[15];
  const float* outproj_w= (const float*)d_in[16];
  const float* fconv_w  = (const float*)d_in[17];
  const float* fconv_b  = (const float*)d_in[18];
  float* out = (float*)d_out;

  const long long XZB  = 2LL * DIN * LSEQ;         // per-batch xz stride
  const long long ROWS = (long long)BN * DIN * LSEQ;

  float* ws = (float*)d_ws;
  size_t o = 0;
  float* xz      = ws + o; o += (size_t)BN * XZB;
  float* xzr     = ws + o; o += (size_t)BN * XZB;
  float* xconv   = ws + o; o += (size_t)ROWS;
  float* xdbl    = ws + o; o += (size_t)BN * (DRK + 2 * DST) * LSEQ;
  float* dpre    = ws + o; o += (size_t)ROWS;
  float* yf      = ws + o; o += (size_t)ROWS;
  float* yb      = ws + o; o += (size_t)ROWS;
  float* fre     = ws + o; o += (size_t)ROWS;
  float* fim     = ws + o; o += (size_t)ROWS;
  float* xcpad   = ws + o; o += (size_t)BN * 2 * DIN * (LSEQ + 4);
  float* convout = ws + o; o += (size_t)BN * 2 * DIN * LSEQ;
  float* zmaxb   = ws + o; o += (size_t)BN * DIN;
  float* ytot    = ws + o; o += (size_t)ROWS;
  // optional packed freq-conv weights [tap][o][i]
  const size_t WPACK = 5ull * (2 * DIN) * (2 * DIN);
  float* wpack   = ws + o;
  const int do_pack = (ws_size >= (o + WPACK) * sizeof(float)) ? 1 : 0;

  // 1. xz[b,e,l] = in_proj_w[e,:] . hidden[b,l,:]
  launch_gemm(stream, inproj_w, hidden, xz,
              2 * DIN, LSEQ, DMOD, 0, (long long)LSEQ * DMOD, XZB,
              DMOD, 1, 1, DMOD, LSEQ, 1, 0);

  // 2. reversed copy for backward branch
  {
    int total = (int)(BN * XZB);
    reverse_l<<<(total + 255) / 256, 256, 0, stream>>>(xz, xzr, total);
  }

  // optional one-time weight repack for the 5-tap frequency conv
  if (do_pack) {
    int total = (int)WPACK;
    pack_w<<<(total + 255) / 256, 256, 0, stream>>>(fconv_w, wpack, total);
  }

  // ---- Mamba branches (fwd then bwd, sharing temporaries) ----
  const float* srcs[2] = { xz, xzr };
  const float* cw[2]   = { conv_w, convb_w };
  const float* cb[2]   = { conv_b, convb_b };
  const float* xpw[2]  = { xproj_w, xprojb_w };
  const float* dpw[2]  = { dtproj_w, dtprojb_w };
  const float* dpb[2]  = { dtproj_b, dtprojb_b };
  const float* al[2]   = { A_log, Ab_log };
  const float* dd[2]   = { Dp, Dbp };
  float*       ybuf[2] = { yf, yb };

  for (int br = 0; br < 2; ++br) {
    int total = (int)ROWS;
    dwconv_silu<<<(total + 255) / 256, 256, 0, stream>>>(srcs[br], cw[br], cb[br], xconv, total);

    // x_dbl[b,e,l] = x_proj_w[e,:] . xconv[b,:,l]   (M=96, K=2048)
    launch_gemm(stream, xpw[br], xconv, xdbl,
                DRK + 2 * DST, LSEQ, DIN,
                0, (long long)DIN * LSEQ, (long long)(DRK + 2 * DST) * LSEQ,
                DIN, 1, LSEQ, 1, LSEQ, 1, 0);

    // delta_pre[b,d,l] = dt_proj_w[d,:] . dt[b,:,l]  (K=64)
    launch_gemm(stream, dpw[br], xdbl, dpre,
                DIN, LSEQ, DRK,
                0, (long long)(DRK + 2 * DST) * LSEQ, (long long)DIN * LSEQ,
                DRK, 1, LSEQ, 1, LSEQ, 1, 0);

    selective_scan<<<(BN * DIN + 127) / 128, 128, 0, stream>>>(
        dpre, dpb[br], al[br], xdbl, xconv, dd[br], srcs[br], ybuf[br]);
  }

  // ---- FFT branch ----
  fft_rows_r2c<<<BN * DIN, 256, 0, stream>>>(xz, fre, fim);
  fft_cols<<<BN * LSEQ, 256, 0, stream>>>(fre, fim, 0);
  {
    int total = BN * 2 * DIN * (LSEQ + 4);
    build_xcpad<<<(total + 255) / 256, 256, 0, stream>>>(fre, fim, xcpad, total);
  }
  // frequency-domain dense conv (K=5): 5 accumulated WMMA GEMMs
  for (int tap = 0; tap < 5; ++tap) {
    const float* Aw  = do_pack ? (wpack + (size_t)tap * (2 * DIN) * (2 * DIN))
                               : (fconv_w + tap);
    int lam = do_pack ? (2 * DIN) : (2 * DIN * 5);
    int lak = do_pack ? 1 : 5;
    launch_gemm(stream, Aw, xcpad + tap, convout,
                2 * DIN, LSEQ, 2 * DIN,
                0, (long long)2 * DIN * (LSEQ + 4), (long long)2 * DIN * LSEQ,
                lam, lak, LSEQ + 4, 1, LSEQ, 1, tap == 0 ? 0 : 1);
  }
  {
    int total = (int)ROWS;
    bias_silu_split<<<(total + 255) / 256, 256, 0, stream>>>(convout, fconv_b, fre, fim, total);
  }
  fft_cols<<<BN * LSEQ, 256, 0, stream>>>(fre, fim, 1);
  fft_rows_c2c<<<BN * DIN, 256, 0, stream>>>(fre, fim, 1);

  // ---- gating, normalization, combination ----
  zmax_kernel<<<(BN * DIN + 127) / 128, 128, 0, stream>>>(xz, zmaxb);
  combine_norm<<<BN * DIN, 256, 0, stream>>>(yf, yb, fre, fim, zmaxb, ytot);

  // 3. out[b,l,o] = out_proj_w[o,:] . ytot[b,:,l]
  launch_gemm(stream, outproj_w, ytot, out,
              DMOD, LSEQ, DIN,
              0, (long long)DIN * LSEQ, (long long)LSEQ * DMOD,
              DIN, 1, LSEQ, 1, 1, DMOD, 0);
}
